// Attention_77025943487081
// MI455X (gfx1250) — matrix operand order repaired and checked by reading, not compiled, not run on hardware
//
#include <hip/hip_runtime.h>

// ---------------------------------------------------------------------------
// Problem constants (from reference): B=4, S=2048, E=1024, H=16, D=64, P=64
// M = B*S = 8192 tokens.
// ---------------------------------------------------------------------------
#define S_LEN 2048
#define EMB   1024
#define NHEAD 16
#define HDIM  64
#define MTOK  8192

typedef __attribute__((ext_vector_type(16))) __bf16 v16bf;
typedef __attribute__((ext_vector_type(8)))  __bf16 v8bf;
typedef __attribute__((ext_vector_type(8)))  float  v8f;
typedef __attribute__((ext_vector_type(4)))  unsigned int v4u;
typedef __attribute__((ext_vector_type(8)))  int v8i;
typedef __attribute__((ext_vector_type(4)))  int v4i;

#define WMMA_BF16(a, b, c) \
  __builtin_amdgcn_wmma_f32_16x16x32_bf16(false, (a), false, (b), (short)0, (c), false, false)

#if defined(__gfx1250__) && __has_builtin(__builtin_amdgcn_tensor_load_to_lds)
#define HAVE_TDM 1
#else
#define HAVE_TDM 0
#endif

// Build a 16-element bf16 fragment from two aligned 16B LDS chunks.
__device__ __forceinline__ v16bf frag2(const __bf16* p0, const __bf16* p1) {
  v8bf a = *(const v8bf*)p0;
  v8bf b = *(const v8bf*)p1;
  return __builtin_shufflevector(a, b, 0, 1, 2, 3, 4, 5, 6, 7, 8, 9, 10, 11, 12, 13, 14, 15);
}

#if HAVE_TDM
// TDM 2D tile load: tile_x contiguous bf16 elements per row, tile_y rows with
// stride_elems between rows, LDS destination padded by pad_amount DWORDs every
// pad_interval DWORDs (encoded per D# spec: interval 3 => 16 DW, amount 3 => 4 DW).
// Issued once per wave (TDM ignores EXEC); caller guards with a wave-uniform branch
// and synchronizes with s_wait_tensorcnt.
__device__ __forceinline__ void tdm_load_2d(unsigned lds_addr, const void* gptr,
                                            unsigned tile_x, unsigned tile_y,
                                            unsigned stride_elems,
                                            unsigned pad_interval, unsigned pad_amount) {
  unsigned long long ga = (unsigned long long)(size_t)gptr;
  v4u g0;
  g0[0] = 1u;                                        // count=1 (valid), user mode
  g0[1] = lds_addr;                                  // LDS byte address
  g0[2] = (unsigned)ga;                              // global addr [31:0]
  g0[3] = (unsigned)((ga >> 32) & 0x01ffffffull) | 0x80000000u;  // [56:32] | type=2
  v8i g1;
  g1[0] = (int)((1u << 16) |                         // data_size = 1 -> 2 bytes
                (1u << 20) |                         // pad_enable
                (pad_interval << 22) | (pad_amount << 25));
  g1[1] = (int)((tile_x & 0xffffu) << 16);           // tensor_dim0 lo16 (== tile_x)
  g1[2] = (int)((tile_x >> 16) | ((tile_y & 0xffffu) << 16));  // dim0 hi / dim1 lo
  g1[3] = (int)((tile_y >> 16) | (tile_x << 16));    // dim1 hi / tile_dim0
  g1[4] = (int)tile_y;                               // tile_dim1, tile_dim2=0
  g1[5] = (int)stride_elems;                         // tensor_dim0_stride lo32
  g1[6] = 0;
  g1[7] = 0;
  v4i z4 = {};
#if __clang_major__ >= 23
  v8i z8 = {};
  __builtin_amdgcn_tensor_load_to_lds(g0, g1, z4, z4, z8, 0);
#else
  __builtin_amdgcn_tensor_load_to_lds(g0, g1, z4, z4, 0);
#endif
}
#endif  // HAVE_TDM

// ---------------------------------------------------------------------------
// Prep kernel 1: effective weights.
//   Wf_v[n*E + k] = W[h,e,d'] * Wh[g,h,d']   with n=(g,d'), k=(h,e)   (v=q,k,v)
//   WcT[n*E + k]  = Wc[k*E + n]
// Stored [n][k] (bf16) so each output column's K-vector is contiguous.
// ---------------------------------------------------------------------------
__global__ __launch_bounds__(256) void prep_weights_kernel(
    const float* __restrict__ Wq, const float* __restrict__ Wqh,
    const float* __restrict__ Wk, const float* __restrict__ Wkh,
    const float* __restrict__ Wv, const float* __restrict__ Wvh,
    const float* __restrict__ Wc,
    __bf16* __restrict__ wfq, __bf16* __restrict__ wfk,
    __bf16* __restrict__ wfv, __bf16* __restrict__ wct) {
  size_t idx = (size_t)blockIdx.x * 256 + threadIdx.x;  // < 4 * E * E
  int which = (int)(idx >> 20);
  size_t r = idx & ((1u << 20) - 1);
  int n = (int)(r >> 10);
  int k = (int)(r & 1023);
  if (which == 3) {
    wct[r] = (__bf16)Wc[(size_t)k * EMB + n];
    return;
  }
  int g = n >> 6, dp = n & 63, h = k >> 6, e = k & 63;
  const float* W  = (which == 0) ? Wq  : (which == 1) ? Wk  : Wv;
  const float* Wh = (which == 0) ? Wqh : (which == 1) ? Wkh : Wvh;
  float w = W[(size_t)h * 4096 + (size_t)e * 64 + dp] *
            Wh[(size_t)g * 1024 + (size_t)h * 64 + dp];
  __bf16* out = (which == 0) ? wfq : (which == 1) ? wfk : wfv;
  out[r] = (__bf16)w;
}

// Prep kernel 2: fused biases bias_v[g,d'] = sum_h b[h,d']*Wh[g,h,d'] + bh[g,d']
__global__ __launch_bounds__(256) void prep_bias_kernel(
    const float* __restrict__ bq, const float* __restrict__ Wqh, const float* __restrict__ bqh,
    const float* __restrict__ bk, const float* __restrict__ Wkh, const float* __restrict__ bkh,
    const float* __restrict__ bv, const float* __restrict__ Wvh, const float* __restrict__ bvh,
    float* __restrict__ biasf) {
  int idx = blockIdx.x * 256 + threadIdx.x;  // < 3 * 1024
  int v = idx >> 10;
  int rem = idx & 1023;
  int g = rem >> 6, dp = rem & 63;
  const float* b  = (v == 0) ? bq  : (v == 1) ? bk  : bv;
  const float* Wh = (v == 0) ? Wqh : (v == 1) ? Wkh : Wvh;
  const float* bh = (v == 0) ? bqh : (v == 1) ? bkh : bvh;
  float acc = bh[g * 64 + dp];
#pragma unroll
  for (int h = 0; h < NHEAD; h++)
    acc += b[h * 64 + dp] * Wh[g * 1024 + h * 64 + dp];
  biasf[idx] = acc;
}

// Prep kernel 3: x (f32) -> bf16
__global__ __launch_bounds__(256) void convert_x_kernel(const float* __restrict__ x,
                                                        __bf16* __restrict__ xb) {
  size_t i = (size_t)blockIdx.x * 256 + threadIdx.x;  // < M * E
  xb[i] = (__bf16)x[i];
}

// Prep kernel 4: p[t, v*64+d] = pos[t,:] @ Wp[:, v*64+d] + bp
__global__ __launch_bounds__(256) void pos_proj_kernel(const float* __restrict__ pos,
                                                       const float* __restrict__ Wp,
                                                       const float* __restrict__ bp,
                                                       float* __restrict__ pvec) {
  size_t idx = (size_t)blockIdx.x * 256 + threadIdx.x;  // < M * 192
  size_t t = idx / 192;
  int c = (int)(idx % 192);
  float acc = bp[c];
  const float* pr = pos + t * 64;
#pragma unroll
  for (int j = 0; j < 64; j++) acc += pr[j] * Wp[j * 192 + c];
  pvec[idx] = acc;
}

// ---------------------------------------------------------------------------
// Tiled bf16 WMMA GEMM:  C[M=8192, N=1024] = A[M,K=1024] * Bw^T  (Bw is [N][K])
// Block: 128x128 tile, 256 threads (8 waves, 4x2 wave grid, 32x64 per wave).
// Tiles move via TDM (tensor_load_to_lds) with double buffering; TDM's LDS
// padding (4 DW every 16 DW) produces the 40-element padded row stride.
// variant 0/1/2: q/k/v epilogue (bias + positional, bf16 out in [B,H,S,D]).
// variant 3    : c_proj epilogue (+bc, f32 out [t, n]).
// ---------------------------------------------------------------------------
#define LDT 40  // padded K-stride of LDS tiles (80B rows, 16B aligned)

__global__ __launch_bounds__(256) void gemm_kernel(
    const __bf16* __restrict__ A, const __bf16* __restrict__ Bw,
    float* __restrict__ outf, __bf16* __restrict__ outbf,
    const float* __restrict__ bias, const float* __restrict__ pvec,
    const float* __restrict__ Wpe, const float* __restrict__ bpe, int variant) {
  __shared__ __align__(16) __bf16 As[2][128 * LDT];
  __shared__ __align__(16) __bf16 Bs[2][128 * LDT];

  const int tid = threadIdx.x;
  const int lane = tid & 31;
  const int wave = tid >> 5;
  const int wm = wave >> 1;   // 0..3
  const int wn = wave & 1;    // 0..1
  const int hf = lane >> 4;   // which 16-lane half
  const int l15 = lane & 15;

  const int mbase = blockIdx.y * 128;
  const int nbase = blockIdx.x * 128;

  v8f acc[2][4] = {};

  // 16-bit fragment K-chunk offsets (wave32), contiguous k = 16*half + s, same for A and B:
  const int ka0 = hf * 16;        // A: lanes 0-15 -> K 0..15, lanes 16-31 -> K 16..31
  const int ka1 = ka0 + 8;
  const int kb0 = hf * 16;        // B: lanes 0-15 -> K 0..15, lanes 16-31 -> K 16..31

  int cur = 0;
#if HAVE_TDM
  const __bf16* Abase = A + (size_t)mbase * EMB;
  const __bf16* Bbase = Bw + (size_t)nbase * EMB;
  if (wave == 0) {
    tdm_load_2d((unsigned)(size_t)&As[0][0], Abase, 32, 128, EMB, 3, 3);
    tdm_load_2d((unsigned)(size_t)&Bs[0][0], Bbase, 32, 128, EMB, 3, 3);
    __builtin_amdgcn_s_wait_tensorcnt(0);
  }
  __syncthreads();
#else
  const int ldr = tid >> 1;
  const int ldc = (tid & 1) * 16;
#endif

  for (int kt = 0; kt < EMB; kt += 32) {
#if HAVE_TDM
    const int nxt = cur ^ 1;
    const bool have_next = (kt + 32) < EMB;
    if (have_next && wave == 0) {  // prefetch next tiles via async DMA
      tdm_load_2d((unsigned)(size_t)&As[nxt][0], Abase + kt + 32, 32, 128, EMB, 3, 3);
      tdm_load_2d((unsigned)(size_t)&Bs[nxt][0], Bbase + kt + 32, 32, 128, EMB, 3, 3);
    }
#else
    __syncthreads();
    {
      const __bf16* ga = A + (size_t)(mbase + ldr) * EMB + kt + ldc;
      const __bf16* gb = Bw + (size_t)(nbase + ldr) * EMB + kt + ldc;
      v8bf a0 = *(const v8bf*)ga;
      v8bf a1 = *(const v8bf*)(ga + 8);
      v8bf b0 = *(const v8bf*)gb;
      v8bf b1 = *(const v8bf*)(gb + 8);
      *(v8bf*)&As[0][ldr * LDT + ldc]     = a0;
      *(v8bf*)&As[0][ldr * LDT + ldc + 8] = a1;
      *(v8bf*)&Bs[0][ldr * LDT + ldc]     = b0;
      *(v8bf*)&Bs[0][ldr * LDT + ldc + 8] = b1;
    }
    __syncthreads();
#endif

    v16bf afr[2];
#pragma unroll
    for (int i = 0; i < 2; i++) {
      const __bf16* base = &As[cur][(wm * 32 + i * 16 + l15) * LDT];
      afr[i] = frag2(base + ka0, base + ka1);
    }
#pragma unroll
    for (int j = 0; j < 4; j++) {
      const __bf16* base = &Bs[cur][(wn * 64 + j * 16 + l15) * LDT];
      v16bf bfr = frag2(base + kb0, base + kb0 + 8);
#pragma unroll
      for (int i = 0; i < 2; i++) acc[i][j] = WMMA_BF16(afr[i], bfr, acc[i][j]);
    }

#if HAVE_TDM
    __syncthreads();  // all waves done reading buffer `cur`
    if (have_next && wave == 0) __builtin_amdgcn_s_wait_tensorcnt(0);
    __syncthreads();  // next buffer published
    cur = nxt;
#endif
  }

  // Epilogue. C-tile layout: VGPR r -> row r (lanes 0-15) / r+8 (lanes 16-31),
  // column = lane&15.
#pragma unroll
  for (int i = 0; i < 2; i++) {
#pragma unroll
    for (int j = 0; j < 4; j++) {
      const int n = nbase + wn * 64 + j * 16 + l15;
#pragma unroll
      for (int r = 0; r < 8; r++) {
        const int m = mbase + wm * 32 + i * 16 + r + hf * 8;  // token index
        float val = acc[i][j][r];
        if (variant < 3) {
          const int g = n >> 6, dp = n & 63;
          val += bias[n] +
                 pvec[(size_t)m * 192 + variant * 64 + dp] * Wpe[variant * 16 + g] +
                 bpe[g];
          const int b_ = m >> 11, s_ = m & (S_LEN - 1);
          // q/k/v stored [B, H, S, D] (bf16)
          outbf[(((size_t)(b_ * NHEAD + g)) * S_LEN + s_) * HDIM + dp] = (__bf16)val;
        } else {
          outf[(size_t)m * EMB + n] = val + bias[n];
        }
      }
    }
  }
}

// ---------------------------------------------------------------------------
// Flash attention (causal, scale 1/8, masked bias -10000 like the reference).
// Q/K/V in [B,H,S,D] bf16. Block = 128 query rows for one (b,h); 8 waves,
// each wave owns 16 query rows. Key tiles of 32. K tile arrives via TDM
// (padded to the 72-element row stride), V is transposed manually.
// Output O in [B,S,H,D] bf16 (== [token, E] for the final GEMM).
// ---------------------------------------------------------------------------
__global__ __launch_bounds__(256) void flash_kernel(const __bf16* __restrict__ Qb,
                                                    const __bf16* __restrict__ Kb,
                                                    const __bf16* __restrict__ Vb,
                                                    __bf16* __restrict__ Ob) {
  __shared__ __align__(16) __bf16 Qs[128 * 72];      // [qrow][d]  (pad 64->72)
  __shared__ __align__(16) __bf16 Ks[32 * 72];       // [key][d]
  __shared__ __align__(16) __bf16 Vts[64 * LDT];     // [d][key]   (pad 32->40)
  __shared__ __align__(16) __bf16 Ps[8 * 16 * LDT];  // per-wave P staging 16x32

  const int tid = threadIdx.x;
  const int lane = tid & 31;
  const int wave = tid >> 5;
  const int hf = lane >> 4;
  const int l15 = lane & 15;

  const int qbase = blockIdx.x * 128;
  const int bh = blockIdx.y;  // b*H + h
  const size_t seqoff = (size_t)bh * S_LEN * HDIM;

  // Load Q tile once.
  {
    const int r = tid >> 1;
    const int c = (tid & 1) * 32;
    const __bf16* g = Qb + seqoff + (size_t)(qbase + r) * HDIM + c;
#pragma unroll
    for (int i = 0; i < 4; i++) *(v8bf*)&Qs[r * 72 + c + i * 8] = *(const v8bf*)(g + i * 8);
  }

  const int qrow = wave * 16;
  v8f o[4] = {};
  float mrun[8], lrun[8];
#pragma unroll
  for (int r = 0; r < 8; r++) { mrun[r] = -__builtin_inff(); lrun[r] = 0.0f; }

  const int ka0 = hf * 16;
  const int ka1 = ka0 + 8;
  const int kb0 = hf * 16;

  __syncthreads();
  // Q A-fragments persist across the key loop (d chunks 0-31 and 32-63).
  v16bf aq0, aq1;
  {
    const __bf16* base = &Qs[(qrow + l15) * 72];
    aq0 = frag2(base + ka0, base + ka1);
    aq1 = frag2(base + 32 + ka0, base + 32 + ka1);
  }

  const int ktmax = (qbase + 127) >> 5;  // inclusive causal limit
  for (int kt = 0; kt <= ktmax; kt++) {
    const int kstart = kt * 32;
    __syncthreads();  // previous iteration fully done with Ks/Vts/Ps
#if HAVE_TDM
    if (wave == 0) {  // K tile 32x64 -> LDS rows padded 64->72 (4 DW per 32 DW)
      tdm_load_2d((unsigned)(size_t)&Ks[0],
                  Kb + seqoff + (size_t)kstart * HDIM, 64, 32, HDIM, 4, 3);
    }
#endif
    {
      const int r = tid >> 3;        // key 0..31
      const int c = (tid & 7) * 8;   // d offset
#if !HAVE_TDM
      const __bf16* gk = Kb + seqoff + (size_t)(kstart + r) * HDIM + c;
      *(v8bf*)&Ks[r * 72 + c] = *(const v8bf*)gk;
#endif
      const __bf16* gv = Vb + seqoff + (size_t)(kstart + r) * HDIM + c;
      v8bf vv = *(const v8bf*)gv;
#pragma unroll
      for (int i = 0; i < 8; i++) Vts[(c + i) * LDT + r] = vv[i];  // transpose V
      if (kt < ktmax) __builtin_prefetch(gv + 32 * HDIM, 0, 3);
    }
#if HAVE_TDM
    if (wave == 0) __builtin_amdgcn_s_wait_tensorcnt(0);
#endif
    __syncthreads();

    // scores: S = Q(16x64) * K^T(64x32) -> two 16x16 f32 tiles
    v8f s0 = {}, s1 = {};
    {
      const __bf16* bb0 = &Ks[l15 * 72];         // keys 0..15
      const __bf16* bb1 = &Ks[(16 + l15) * 72];  // keys 16..31
      v16bf k00 = frag2(bb0 + kb0, bb0 + kb0 + 8);
      v16bf k01 = frag2(bb0 + 32 + kb0, bb0 + 32 + kb0 + 8);
      v16bf k10 = frag2(bb1 + kb0, bb1 + kb0 + 8);
      v16bf k11 = frag2(bb1 + 32 + kb0, bb1 + 32 + kb0 + 8);
      s0 = WMMA_BF16(aq0, k00, s0);
      s0 = WMMA_BF16(aq1, k01, s0);
      s1 = WMMA_BF16(aq0, k10, s1);
      s1 = WMMA_BF16(aq1, k11, s1);
    }

    // online softmax (row reductions stay within each 16-lane half)
    float p0[8], p1[8], alpha[8];
#pragma unroll
    for (int r = 0; r < 8; r++) {
      const int m = qbase + qrow + r + hf * 8;
      const int c0 = kstart + l15;
      const int c1 = c0 + 16;
      float x0 = s0[r] * 0.125f;
      float x1 = s1[r] * 0.125f;
      if (c0 > m) x0 = -10000.0f;
      if (c1 > m) x1 = -10000.0f;
      float rm = fmaxf(x0, x1);
      rm = fmaxf(rm, __shfl_xor(rm, 1));
      rm = fmaxf(rm, __shfl_xor(rm, 2));
      rm = fmaxf(rm, __shfl_xor(rm, 4));
      rm = fmaxf(rm, __shfl_xor(rm, 8));
      float mnew = fmaxf(mrun[r], rm);
      alpha[r] = __expf(mrun[r] - mnew);
      mrun[r] = mnew;
      p0[r] = __expf(x0 - mnew);
      p1[r] = __expf(x1 - mnew);
      float rs = p0[r] + p1[r];
      rs += __shfl_xor(rs, 1);
      rs += __shfl_xor(rs, 2);
      rs += __shfl_xor(rs, 4);
      rs += __shfl_xor(rs, 8);
      lrun[r] = lrun[r] * alpha[r] + rs;
    }

#pragma unroll
    for (int j = 0; j < 4; j++)
#pragma unroll
      for (int r = 0; r < 8; r++) o[j][r] *= alpha[r];

    // stage P (C layout) through LDS, reload as A fragment
    {
      __bf16* pw = &Ps[wave * 16 * LDT];
#pragma unroll
      for (int r = 0; r < 8; r++) {
        const int row = r + hf * 8;
        pw[row * LDT + l15]      = (__bf16)p0[r];
        pw[row * LDT + 16 + l15] = (__bf16)p1[r];
      }
    }
    __syncthreads();
    v16bf pf;
    {
      const __bf16* base = &Ps[wave * 16 * LDT + l15 * LDT];
      pf = frag2(base + ka0, base + ka1);
    }
#pragma unroll
    for (int j = 0; j < 4; j++) {
      const __bf16* vb_ = &Vts[(j * 16 + l15) * LDT];
      v16bf vf = frag2(vb_ + kb0, vb_ + kb0 + 8);
      o[j] = WMMA_BF16(pf, vf, o[j]);
    }
  }

  // epilogue: normalize and write O in [B, S, H, D] (== [token, E]) bf16
  const int b_ = bh >> 4, h_ = bh & 15;
#pragma unroll
  for (int j = 0; j < 4; j++) {
#pragma unroll
    for (int r = 0; r < 8; r++) {
      const float inv = 1.0f / lrun[r];
      const int srow = qbase + qrow + r + hf * 8;
      const int d = j * 16 + l15;
      Ob[(((size_t)b_ * S_LEN + srow) * NHEAD + h_) * HDIM + d] = (__bf16)(o[j][r] * inv);
    }
  }
}

// ---------------------------------------------------------------------------
// Host side
// ---------------------------------------------------------------------------
extern "C" void kernel_launch(void* const* d_in, const int* in_sizes, int n_in,
                              void* d_out, int out_size, void* d_ws, size_t ws_size,
                              hipStream_t stream) {
  const float* hidden = (const float*)d_in[0];
  const float* pos    = (const float*)d_in[1];
  const float* Wq  = (const float*)d_in[2];  const float* bq  = (const float*)d_in[3];
  const float* Wk  = (const float*)d_in[4];  const float* bk  = (const float*)d_in[5];
  const float* Wv  = (const float*)d_in[6];  const float* bv  = (const float*)d_in[7];
  const float* Wqh = (const float*)d_in[8];  const float* bqh = (const float*)d_in[9];
  const float* Wkh = (const float*)d_in[10]; const float* bkh = (const float*)d_in[11];
  const float* Wvh = (const float*)d_in[12]; const float* bvh = (const float*)d_in[13];
  const float* Wp  = (const float*)d_in[14]; const float* bp  = (const float*)d_in[15];
  const float* Wpe = (const float*)d_in[16]; const float* bpe = (const float*)d_in[17];
  const float* Wc  = (const float*)d_in[18]; const float* bc  = (const float*)d_in[19];

  char* ws = (char*)d_ws;
  const size_t EE2  = (size_t)EMB * EMB * 2;        // 2 MiB per bf16 E x E weight
  const size_t ME2  = (size_t)MTOK * EMB * 2;       // 16 MiB per bf16 token matrix
  size_t off = 0;
  __bf16* wfq = (__bf16*)(ws + off); off += EE2;
  __bf16* wfk = (__bf16*)(ws + off); off += EE2;
  __bf16* wfv = (__bf16*)(ws + off); off += EE2;
  __bf16* wct = (__bf16*)(ws + off); off += EE2;
  float* biasf = (float*)(ws + off); off += 16384;           // 3*1024 f32 (padded)
  float* pvec  = (float*)(ws + off); off += (size_t)MTOK * 192 * 4;
  __bf16* xbf = (__bf16*)(ws + off); off += ME2;
  __bf16* qb  = (__bf16*)(ws + off); off += ME2;
  __bf16* kb  = (__bf16*)(ws + off); off += ME2;
  __bf16* vb  = (__bf16*)(ws + off); off += ME2;
  __bf16* ob  = (__bf16*)(ws + off); off += ME2;
  (void)ws_size; (void)in_sizes; (void)n_in; (void)out_size;

  // 1) prep
  prep_weights_kernel<<<(4 * EMB * EMB) / 256, 256, 0, stream>>>(
      Wq, Wqh, Wk, Wkh, Wv, Wvh, Wc, wfq, wfk, wfv, wct);
  prep_bias_kernel<<<12, 256, 0, stream>>>(bq, Wqh, bqh, bk, Wkh, bkh, bv, Wvh, bvh, biasf);
  convert_x_kernel<<<((size_t)MTOK * EMB) / 256, 256, 0, stream>>>(hidden, xbf);
  pos_proj_kernel<<<((size_t)MTOK * 192) / 256, 256, 0, stream>>>(pos, Wp, bp, pvec);

  // 2) q/k/v projections (fused effective GEMMs)
  dim3 gg(EMB / 128, MTOK / 128);
  gemm_kernel<<<gg, 256, 0, stream>>>(xbf, wfq, nullptr, qb, biasf + 0,    pvec, Wpe, bpe, 0);
  gemm_kernel<<<gg, 256, 0, stream>>>(xbf, wfk, nullptr, kb, biasf + 1024, pvec, Wpe, bpe, 1);
  gemm_kernel<<<gg, 256, 0, stream>>>(xbf, wfv, nullptr, vb, biasf + 2048, pvec, Wpe, bpe, 2);

  // 3) causal flash attention per (b, h)
  flash_kernel<<<dim3(S_LEN / 128, 4 * NHEAD), 256, 0, stream>>>(qb, kb, vb, ob);

  // 4) output projection -> f32 d_out
  gemm_kernel<<<gg, 256, 0, stream>>>(ob, wct, (float*)d_out, nullptr, bc,
                                      nullptr, nullptr, nullptr, 3);
}
